// Dcls2d_91156385890466
// MI455X (gfx1250) — matrix operand order repaired and checked by reading, not compiled, not run on hardware
//
#include <hip/hip_runtime.h>

typedef __attribute__((ext_vector_type(8)))  float  v8f;
typedef __attribute__((ext_vector_type(8)))  __bf16 v8bf;
typedef __attribute__((ext_vector_type(16))) __bf16 v16bf;

#define KH 7
#define KW 7
#define PADW 3
#define N_IMG 32
#define C_IN 64
#define C_OUT 128
#define HW 64
#define KSP 26
#define IC_CHUNK 32
#define COLS 70            // 64 output cols + 6 halo
#define COL_BF 40          // 32 ch (64B) + 8 bf16 pad -> 80B col stride
#define PLANE (KH * COLS * COL_BF) // bf16 elements per LDS plane (19600)
#define A_PLANE (4 * C_OUT * 8)    // bf16 elements per A hi/lo plane (4096)

union Frag16 { v16bf v; v8bf h[2]; };

// ---------------------------------------------------------------------------
// Kernel 1: bilinear construction of dense kernel, split into bf16 hi/lo and
// packed to match the 16-bit WMMA A-fragment layout (ISA 7.12.2):
//   KtB[cell][chunk][plane][run][o][8]   run r = channels [8r, 8r+8)
// Lane (m, g=lane>=16) loads runs g and g+2 as two contiguous 16B chunks.
// ---------------------------------------------------------------------------
__global__ void dcls_build_kernel(const float* __restrict__ weight,
                                  const float* __restrict__ P,
                                  __bf16* __restrict__ KtB) {
    int tid = blockIdx.x * blockDim.x + threadIdx.x;
    if (tid >= C_OUT * C_IN) return;
    int o = tid / C_IN;
    int i = tid - o * C_IN;

    const float* w  = weight + (size_t)(o * C_IN + i) * KSP;
    const float* P1 = P + ((size_t)(0 * C_OUT + o) * C_IN + i) * KSP; // kw axis
    const float* P2 = P + ((size_t)(1 * C_OUT + o) * C_IN + i) * KSP; // kh axis

    float acc[KH * KW];
#pragma unroll
    for (int c = 0; c < KH * KW; ++c) acc[c] = 0.f;

    for (int k = 0; k < KSP; ++k) {
        float wv = w[k];
        float p1 = P1[k] + (float)(KW / 2);
        float p2 = P2[k] + (float)(KH / 2);
        float f1 = floorf(p1), f2 = floorf(p2);
        int   i1 = (int)f1,    i2 = (int)f2;
        float r1 = p1 - f1,    r2 = p2 - f2;
#pragma unroll
        for (int kh = 0; kh < KH; ++kh) {
            float ch = (kh == i2) ? (1.f - r2) : ((kh == i2 + 1) ? r2 : 0.f);
#pragma unroll
            for (int kw = 0; kw < KW; ++kw) {
                float cw = (kw == i1) ? (1.f - r1) : ((kw == i1 + 1) ? r1 : 0.f);
                acc[kh * KW + kw] += wv * (cw * ch);
            }
        }
    }

    int chunk = i >> 5;          // which 32-channel K block
    int run   = (i >> 3) & 3;    // 8-channel run within block
    int pos   = i & 7;
#pragma unroll
    for (int cell = 0; cell < KH * KW; ++cell) {
        float  v  = acc[cell];
        __bf16 hi = (__bf16)v;
        __bf16 lo = (__bf16)(v - (float)hi);
        size_t base = ((size_t)(cell * 2 + chunk) * 2) * (size_t)A_PLANE;
        size_t off  = ((size_t)run * C_OUT + o) * 8 + pos;
        KtB[base + off]           = hi;   // plane 0 = hi
        KtB[base + A_PLANE + off] = lo;   // plane 1 = lo
    }
}

// ---------------------------------------------------------------------------
// Kernel 2: implicit-GEMM conv on V_WMMA_F32_16X16X32_BF16 with bf16x3 split
// (Ah*Bh + Ah*Bl + Al*Bh, fp32 accumulate -> ~1e-5 relative accuracy).
// Block = one (n,h) output row; 8 waves x (16 oc x 64 cols).
// ---------------------------------------------------------------------------
__global__ void __launch_bounds__(256) dcls_conv_wmma(
    const float* __restrict__ x, const __bf16* __restrict__ KtB,
    const float* __restrict__ bias, float* __restrict__ out) {
    extern __shared__ __bf16 xsb[]; // [2][KH][COLS][COL_BF]
    __bf16* xh = xsb;
    __bf16* xl = xsb + PLANE;

    int blk  = blockIdx.x;     // 0 .. N_IMG*HW-1
    int n    = blk / HW;
    int h    = blk - n * HW;
    int tid  = threadIdx.x;
    int lane = tid & 31;       // wave32
    int wv   = tid >> 5;       // 0..7
    int obase = wv * 16;
    int m    = lane & 15;      // A: M row / B,C: N column
    int g    = lane >> 4;      // lane-group selector (0/1)

    v8f acc0 = {}, acc1 = {}, acc2 = {}, acc3 = {};

    for (int icb = 0; icb < C_IN; icb += IC_CHUNK) {
        int chunk = icb >> 5;
        __syncthreads(); // protect previous chunk's LDS reads
        // stage x[n, icb..icb+31, h-3..h+3, -3..66], split to bf16 hi/lo,
        // store channel-fastest so a B fragment is 2 contiguous 16B reads
        for (int idx = tid; idx < IC_CHUNK * KH * COLS; idx += 256) {
            int ch  = idx / (KH * COLS);
            int rem = idx - ch * (KH * COLS);
            int r   = rem / COLS;
            int c   = rem - r * COLS;
            int hy  = h + r - PADW;
            int xx  = c - PADW;
            float v = 0.f;
            if ((unsigned)hy < HW && (unsigned)xx < HW)
                v = x[(((size_t)n * C_IN + (icb + ch)) * HW + hy) * HW + xx];
            __bf16 hi = (__bf16)v;
            __bf16 lo = (__bf16)(v - (float)hi);
            int la = (r * COLS + c) * COL_BF + ch;
            xh[la] = hi;
            xl[la] = lo;
        }
        __syncthreads();

        for (int kh = 0; kh < KH; ++kh) {
            for (int kw = 0; kw < KW; ++kw) {
                int cell = kh * KW + kw;
                // ---- A fragments (global, L2-resident, coalesced 16B/lane)
                const __bf16* ab =
                    KtB + ((size_t)(cell * 2 + chunk) * 2) * (size_t)A_PLANE;
                int aoff1 = ((g)     * C_OUT + obase + m) * 8;
                int aoff2 = ((g + 2) * C_OUT + obase + m) * 8;
                Frag16 ahi, alo;
                ahi.h[0] = *(const v8bf*)(ab + aoff1);
                ahi.h[1] = *(const v8bf*)(ab + aoff2);
                alo.h[0] = *(const v8bf*)(ab + A_PLANE + aoff1);
                alo.h[1] = *(const v8bf*)(ab + A_PLANE + aoff2);
                // ---- B fragments (LDS): lane-group g reads ch [8g,8g+8) and [16+8g,24+8g)
                int bbase = (kh * COLS + m + kw) * COL_BF + 8 * g;
#pragma unroll
                for (int t = 0; t < 4; ++t) {
                    int bo = bbase + t * 16 * COL_BF;
                    Frag16 bhi, blo;
                    bhi.h[0] = *(const v8bf*)(xh + bo);
                    bhi.h[1] = *(const v8bf*)(xh + bo + 16);
                    blo.h[0] = *(const v8bf*)(xl + bo);
                    blo.h[1] = *(const v8bf*)(xl + bo + 16);
                    v8f* acc = (t == 0) ? &acc0 : (t == 1) ? &acc1
                             : (t == 2) ? &acc2 : &acc3;
                    *acc = __builtin_amdgcn_wmma_f32_16x16x32_bf16(
                        false, ahi.v, false, bhi.v, (short)0, *acc, false, false);
                    *acc = __builtin_amdgcn_wmma_f32_16x16x32_bf16(
                        false, ahi.v, false, blo.v, (short)0, *acc, false, false);
                    *acc = __builtin_amdgcn_wmma_f32_16x16x32_bf16(
                        false, alo.v, false, bhi.v, (short)0, *acc, false, false);
                }
            }
        }
    }

    // C layout: VGPR r -> M = r + 8*g ; N = m; sub-tile t -> cols t*16+m
#pragma unroll
    for (int r = 0; r < 8; ++r) {
        int o = obase + r + 8 * g;
        float bv = bias[o];
        float* op = out + (((size_t)n * C_OUT + o) * HW + h) * HW + m;
        op[0]  = acc0[r] + bv;
        op[16] = acc1[r] + bv;
        op[32] = acc2[r] + bv;
        op[48] = acc3[r] + bv;
    }
}

extern "C" void kernel_launch(void* const* d_in, const int* in_sizes, int n_in,
                              void* d_out, int out_size, void* d_ws, size_t ws_size,
                              hipStream_t stream) {
    const float* x      = (const float*)d_in[0];
    const float* weight = (const float*)d_in[1];
    const float* P      = (const float*)d_in[2];
    const float* bias   = (const float*)d_in[3];
    float*  out = (float*)d_out;
    __bf16* KtB = (__bf16*)d_ws; // 49*2*2*4*128*8 bf16 = 1.6 MB packed kernel

    dcls_build_kernel<<<(C_OUT * C_IN + 255) / 256, 256, 0, stream>>>(
        weight, P, KtB);

    dcls_conv_wmma<<<N_IMG * HW, 256, 2 * PLANE * sizeof(__bf16), stream>>>(
        x, KtB, bias, out);
}
